// Qwen3Attention_77300821393694
// MI455X (gfx1250) — matrix operand order repaired and checked by reading, not compiled, not run on hardware
//
#include <hip/hip_runtime.h>
#include <hip/hip_bf16.h>

// ---------------------------------------------------------------------------
// Qwen3 attention block for MI455X (gfx1250, wave32, WMMA).
// All GEMMs use v_wmma_f32_16x16x32_bf16 (bf16 operands, f32 accum).
// ---------------------------------------------------------------------------

typedef __attribute__((ext_vector_type(16))) __bf16 v16bf;
typedef __attribute__((ext_vector_type(8)))  __bf16 v8bf;
typedef __attribute__((ext_vector_type(8)))  float  v8f;

union V16 {
    v16bf v;
    v8bf  h[2];
    __bf16 b[16];
};

static constexpr int BATCH  = 2;
static constexpr int SEQ    = 2048;
static constexpr int HIDDEN = 2048;
static constexpr int NH     = 16;
static constexpr int NKV    = 8;
static constexpr int HD     = 128;
static constexpr int QDIM   = NH * HD;   // 2048
static constexpr int KVDIM  = NKV * HD;  // 1024
static constexpr int NTOK   = BATCH * SEQ; // 4096

__device__ inline v8f wmma_bf16(v16bf a, v16bf b, v8f c) {
    // D = A(16x32) * B(32x16) + C, f32 accumulate
    return __builtin_amdgcn_wmma_f32_16x16x32_bf16(
        false, a, false, b, (short)0, c, false, false);
}

// A-frag (16x32 bf16, M rows x K): lane L<16 holds row M=L, K = k0+0..7 (h[0])
// and k0+16..23 (h[1]); lane L>=16 holds row M=L-16, K = k0+8..15 / k0+24..31.
__device__ inline v16bf load_a_frag(const __bf16* __restrict__ A, int lda,
                                    int row0, int k0) {
    const int lane = threadIdx.x & 31;
    const int r    = row0 + (lane & 15);
    const int kk   = k0 + ((lane >> 4) << 3);
    const __bf16* p = A + (long)r * lda + kk;
    V16 u;
    u.h[0] = *(const v8bf*)(p);
    u.h[1] = *(const v8bf*)(p + 16);
    return u.v;
}

// B-frag (32x16 bf16, K x N) sourced from an NT (row-major N x K) matrix:
// lane n (n = lane&15) holds column N=col0+n, K = k0 + (lane>>4)*8 + 0..7 (h[0])
// and k0 + 16 + (lane>>4)*8 + 0..7 (h[1]) of row (col0+n) of the NT matrix.
__device__ inline v16bf load_b_frag_nt(const __bf16* __restrict__ Bm, int ldb,
                                       int col0, int k0) {
    const int lane = threadIdx.x & 31;
    const int c    = col0 + (lane & 15);
    const int kk   = k0 + ((lane >> 4) << 3);
    const __bf16* p = Bm + (long)c * ldb + kk;
    V16 u;
    u.h[0] = *(const v8bf*)(p);
    u.h[1] = *(const v8bf*)(p + 16);
    return u.v;
}

// ---------------------------------------------------------------------------
// fp32 -> bf16 pack
// ---------------------------------------------------------------------------
__global__ void cvt_f32_bf16_kernel(const float* __restrict__ src,
                                    __bf16* __restrict__ dst, long n) {
    long i = (long)blockIdx.x * blockDim.x + threadIdx.x;
    if (i < n) dst[i] = (__bf16)src[i];
}

// ---------------------------------------------------------------------------
// GEMM: C(MxN, f32) = A(MxK, bf16 row-major) * B(NxK, bf16 row-major)^T
// block = 256 threads = 8 waves; block tile 64(M) x 128(N); wave tile 32x32.
// ---------------------------------------------------------------------------
__global__ __launch_bounds__(256)
void gemm_bf16_nt_kernel(const __bf16* __restrict__ A,
                         const __bf16* __restrict__ B,
                         float* __restrict__ C, int M, int N, int K) {
    const int wave = threadIdx.x >> 5;
    const int wm   = wave >> 2;          // 0..1
    const int wn   = wave & 3;           // 0..3
    const int m0   = blockIdx.y * 64 + wm * 32;
    const int n0   = blockIdx.x * 128 + wn * 32;
    const int lane = threadIdx.x & 31;

    const v8f vzero = {0.f, 0.f, 0.f, 0.f, 0.f, 0.f, 0.f, 0.f};
    v8f acc[2][2];
    acc[0][0] = vzero; acc[0][1] = vzero; acc[1][0] = vzero; acc[1][1] = vzero;

    for (int k0 = 0; k0 < K; k0 += 32) {
        // prefetch streaming weight operand ~2 iterations ahead
        if (k0 + 64 < K) {
            __builtin_prefetch(B + (long)(n0 + (lane & 15)) * K + k0 + 64, 0, 1);
        }
        v16bf a0 = load_a_frag(A, K, m0,      k0);
        v16bf a1 = load_a_frag(A, K, m0 + 16, k0);
        v16bf b0 = load_b_frag_nt(B, K, n0,      k0);
        v16bf b1 = load_b_frag_nt(B, K, n0 + 16, k0);
        acc[0][0] = wmma_bf16(a0, b0, acc[0][0]);
        acc[0][1] = wmma_bf16(a0, b1, acc[0][1]);
        acc[1][0] = wmma_bf16(a1, b0, acc[1][0]);
        acc[1][1] = wmma_bf16(a1, b1, acc[1][1]);
    }

    // C/D layout: VGPR v, lanes 0-15 -> row m0+v col lane; lanes 16-31 -> row m0+8+v
    const int rbase = (lane >> 4) * 8;
    const int cbase = lane & 15;
#pragma unroll
    for (int i = 0; i < 2; ++i)
#pragma unroll
        for (int j = 0; j < 2; ++j)
#pragma unroll
            for (int v = 0; v < 8; ++v) {
                int row = m0 + i * 16 + rbase + v;
                int col = n0 + j * 16 + cbase;
                C[(long)row * N + col] = acc[i][j][v];
            }
}

// ---------------------------------------------------------------------------
// Per-(token, head-slot) RMSNorm + RoPE, fp32 in -> bf16 out.
// Slot 0..15 = q heads, 16..23 = k heads, 24..31 = v heads (convert only).
// One block (128 threads = 4 waves) per (token, slot); thread d = dim index.
// ---------------------------------------------------------------------------
__global__ __launch_bounds__(128)
void norm_rope_kernel(const float* __restrict__ Qf, const float* __restrict__ Kf,
                      const float* __restrict__ Vf,
                      const float* __restrict__ cosb, const float* __restrict__ sinb,
                      const float* __restrict__ qw, const float* __restrict__ kw,
                      __bf16* __restrict__ Qb, __bf16* __restrict__ Kb,
                      __bf16* __restrict__ Vb) {
    const int slot = blockIdx.x & 31;
    const long tok = blockIdx.x >> 5;       // b*SEQ + s
    const int d    = threadIdx.x;           // 0..127

    const float* src; __bf16* dst; const float* w;
    if (slot < 16) {
        src = Qf + tok * QDIM  + slot * HD;
        dst = Qb + tok * QDIM  + slot * HD;
        w = qw;
    } else if (slot < 24) {
        src = Kf + tok * KVDIM + (slot - 16) * HD;
        dst = Kb + tok * KVDIM + (slot - 16) * HD;
        w = kw;
    } else {
        long o = tok * KVDIM + (slot - 24) * HD;
        Vb[o + d] = (__bf16)Vf[o + d];      // uniform per-block path
        return;
    }

    __shared__ float red[4];
    float x = src[d];
    float s = x * x;
#pragma unroll
    for (int m = 16; m >= 1; m >>= 1) s += __shfl_xor(s, m, 32);
    if ((threadIdx.x & 31) == 0) red[threadIdx.x >> 5] = s;
    __syncthreads();
    float var = (red[0] + red[1] + red[2] + red[3]) * (1.0f / HD);
    float rs  = rsqrtf(var + 1e-6f);

    const int pd  = (d + 64) & 127;
    float xn   = x * rs * w[d];
    float xpn  = src[pd] * rs * w[pd];
    float rot  = (d < 64) ? -xpn : xpn;
    float c    = cosb[tok * HD + d];
    float sn   = sinb[tok * HD + d];
    dst[d] = (__bf16)(xn * c + rot * sn);
}

// ---------------------------------------------------------------------------
// Fused causal attention. One block (256 thr = 8 waves) per (b,h, 16-row qtile).
// Scores live entirely in VGPRs (wave w owns key tiles [16w, 16w+16) -> up to
// 16 v8f accumulators/lane). Softmax: in-register + shfl_xor(w=16) + small LDS
// cross-wave reduce. Probs stream to d_out (f32) incl. zero tail. PV: per-wave
// LDS transpose staging (C-layout -> A-frag) + WMMA against gathered V B-frags,
// cross-wave f32 reduction via LDS atomics, bf16 ctx out.
// ---------------------------------------------------------------------------
__global__ __launch_bounds__(256)
void attn_kernel(const __bf16* __restrict__ Qb, const __bf16* __restrict__ Kb,
                 const __bf16* __restrict__ Vb,
                 float* __restrict__ attn_out, __bf16* __restrict__ ctx) {
    const int qt   = blockIdx.x;            // 0..SEQ/16-1
    const int bh   = blockIdx.y;            // b*NH + h
    const int b    = bh >> 4;
    const int h    = bh & 15;
    const int kvh  = h >> 1;                // NH/NKV = 2
    const int q0   = qt * 16;
    const int lane = threadIdx.x & 31;
    const int wave = threadIdx.x >> 5;      // 0..7
    const int half = lane >> 4;
    const int lc   = lane & 15;
    const int kvlim = q0 + 16;              // keys >= kvlim are fully masked
    const int ntile = kvlim >> 4;

    __shared__ float redbuf[8][16];
    __shared__ float ctxred[16][HD];
    __shared__ __align__(16) __bf16 stage[8][16][32];

    const __bf16* Qp = Qb + (long)b * SEQ * QDIM  + h   * HD;
    const __bf16* Kp = Kb + (long)b * SEQ * KVDIM + kvh * HD;
    const __bf16* Vp = Vb + (long)b * SEQ * KVDIM + kvh * HD;

    // Q A-frags for this tile: 16 rows x 128 (4 K-chunks), reused for all keys
    v16bf qf[4];
#pragma unroll
    for (int kk = 0; kk < 4; ++kk) qf[kk] = load_a_frag(Qp, QDIM, q0, kk * 32);

    const v8f vzero = {0.f, 0.f, 0.f, 0.f, 0.f, 0.f, 0.f, 0.f};
    const float scale = 0.08838834764831845f;   // 1/sqrt(128)
    const float NEG_INF = -__builtin_inff();

    // ---- scores (QK^T), masked + scaled, kept in registers -----------------
    v8f sacc[16];
#pragma unroll
    for (int i = 0; i < 16; ++i) {
        sacc[i] = vzero;
        const int t = wave * 16 + i;
        if (t < ntile) {
            const int kb = t * 16;
            v8f a = vzero;
#pragma unroll
            for (int kk = 0; kk < 4; ++kk)
                a = wmma_bf16(qf[kk], load_b_frag_nt(Kp, KVDIM, kb, kk * 32), a);
#pragma unroll
            for (int v = 0; v < 8; ++v) {
                int r   = half * 8 + v;
                int key = kb + lc;
                float sv = a[v] * scale;
                a[v] = (key > q0 + r) ? NEG_INF : sv;
            }
            sacc[i] = a;
        }
    }

    // ---- row max -----------------------------------------------------------
    float rmax[8];
#pragma unroll
    for (int v = 0; v < 8; ++v) {
        float m = NEG_INF;
#pragma unroll
        for (int i = 0; i < 16; ++i)
            if (wave * 16 + i < ntile) m = fmaxf(m, sacc[i][v]);
#pragma unroll
        for (int msk = 8; msk >= 1; msk >>= 1) m = fmaxf(m, __shfl_xor(m, msk, 16));
        rmax[v] = m;
    }
    if (lc == 0) {
#pragma unroll
        for (int v = 0; v < 8; ++v) redbuf[wave][half * 8 + v] = rmax[v];
    }
    __syncthreads();
#pragma unroll
    for (int v = 0; v < 8; ++v) {
        float m = redbuf[0][half * 8 + v];
#pragma unroll
        for (int w2 = 1; w2 < 8; ++w2) m = fmaxf(m, redbuf[w2][half * 8 + v]);
        rmax[v] = m;
    }
    __syncthreads();

    // ---- exp + row sum -----------------------------------------------------
    float rsum[8];
#pragma unroll
    for (int v = 0; v < 8; ++v) {
        float s = 0.f;
#pragma unroll
        for (int i = 0; i < 16; ++i) {
            if (wave * 16 + i < ntile) {
                float p = __expf(sacc[i][v] - rmax[v]);
                sacc[i][v] = p;
                s += p;
            }
        }
#pragma unroll
        for (int msk = 8; msk >= 1; msk >>= 1) s += __shfl_xor(s, msk, 16);
        rsum[v] = s;
    }
    if (lc == 0) {
#pragma unroll
        for (int v = 0; v < 8; ++v) redbuf[wave][half * 8 + v] = rsum[v];
    }
    __syncthreads();
#pragma unroll
    for (int v = 0; v < 8; ++v) {
        float s = 0.f;
#pragma unroll
        for (int w2 = 0; w2 < 8; ++w2) s += redbuf[w2][half * 8 + v];
        rsum[v] = 1.0f / s;
    }

    // ---- normalize + stream probs to HBM ----------------------------------
    float* arow = attn_out + ((long)bh * SEQ + q0) * SEQ;
#pragma unroll
    for (int i = 0; i < 16; ++i) {
        const int t = wave * 16 + i;
        if (t < ntile) {
#pragma unroll
            for (int v = 0; v < 8; ++v) {
                float p = sacc[i][v] * rsum[v];
                sacc[i][v] = p;
                arow[(long)(half * 8 + v) * SEQ + t * 16 + lc] = p;
            }
        }
    }
    // zero-fill the masked tail [kvlim, SEQ) (128-bit stores)
    {
        const int tail4 = (SEQ - kvlim) >> 2;
        const float4 z = make_float4(0.f, 0.f, 0.f, 0.f);
        for (int idx = threadIdx.x; idx < 16 * tail4; idx += 256) {
            int r  = idx / tail4;
            int c4 = idx - r * tail4;
            *(float4*)(arow + (long)r * SEQ + kvlim + c4 * 4) = z;
        }
    }

    // ---- zero ctx reduction buffer -----------------------------------------
    for (int idx = threadIdx.x; idx < 16 * HD; idx += 256)
        ((float*)ctxred)[idx] = 0.f;
    __syncthreads();

    // ---- PV: wave-private k range [256w, 256w+256), chunks of K=32 ---------
    v8f cacc[8];
#pragma unroll
    for (int n = 0; n < 8; ++n) cacc[n] = vzero;

#pragma unroll
    for (int c = 0; c < 8; ++c) {
        const int tA = wave * 16 + 2 * c;
        if (tA < ntile) {
            // C-layout -> per-wave LDS staging (bf16), zero-padded past ntile
#pragma unroll
            for (int u2 = 0; u2 < 2; ++u2) {
                const int i = 2 * c + u2;
                const bool ok = (wave * 16 + i) < ntile;
#pragma unroll
                for (int v = 0; v < 8; ++v) {
                    float val = ok ? sacc[i][v] : 0.f;
                    stage[wave][half * 8 + v][u2 * 16 + lc] = (__bf16)val;
                }
            }
            // A-frag load from staging (compiler inserts s_wait_dscnt)
            V16 ua;
            const __bf16* sp = &stage[wave][lc][half * 8];
            ua.h[0] = *(const v8bf*)(sp);
            ua.h[1] = *(const v8bf*)(sp + 16);
            const v16bf pa = ua.v;

            const int kc = wave * 256 + c * 32;
#pragma unroll
            for (int n = 0; n < 8; ++n) {
                V16 ub;  // B-frag of V: column d = n*16+lc, K strided gather
                const __bf16* vp = Vp + (long)(kc + half * 8) * KVDIM + n * 16 + lc;
#pragma unroll
                for (int e = 0; e < 16; ++e) ub.b[e] = vp[(long)(e + (e & 8)) * KVDIM];
                cacc[n] = wmma_bf16(pa, ub.v, cacc[n]);
            }
        }
    }

    // ---- cross-wave ctx reduction + bf16 store -----------------------------
#pragma unroll
    for (int n = 0; n < 8; ++n)
#pragma unroll
        for (int v = 0; v < 8; ++v)
            atomicAdd(&ctxred[half * 8 + v][n * 16 + lc], cacc[n][v]);
    __syncthreads();

    for (int idx = threadIdx.x; idx < 16 * HD; idx += 256) {
        int r = idx >> 7, d = idx & 127;
        ctx[((long)b * SEQ + q0 + r) * QDIM + h * HD + d] = (__bf16)ctxred[r][d];
    }
}

// ---------------------------------------------------------------------------
// Host side
// ---------------------------------------------------------------------------
extern "C" void kernel_launch(void* const* d_in, const int* in_sizes, int n_in,
                              void* d_out, int out_size, void* d_ws, size_t ws_size,
                              hipStream_t stream) {
    (void)in_sizes; (void)n_in; (void)out_size; (void)ws_size;

    const float* hidden = (const float*)d_in[0];
    const float* cosb   = (const float*)d_in[1];
    const float* sinb   = (const float*)d_in[2];
    const float* Wq     = (const float*)d_in[3];
    const float* Wk     = (const float*)d_in[4];
    const float* Wv     = (const float*)d_in[5];
    const float* Wo     = (const float*)d_in[6];
    const float* qw     = (const float*)d_in[7];
    const float* kw     = (const float*)d_in[8];

    float* out_proj = (float*)d_out;                                  // (B,S,H)
    float* attn_out = (float*)d_out + (long)NTOK * HIDDEN;            // (B,NH,S,S)

    // Workspace carve-up (bytes, 256B aligned)
    char* ws = (char*)d_ws;
    size_t off = 0;
    auto carve = [&](size_t bytes) {
        char* p = ws + off;
        off += (bytes + 255) & ~(size_t)255;
        return p;
    };
    __bf16* Xb  = (__bf16*)carve((size_t)NTOK * HIDDEN * 2);   // 16 MiB
    __bf16* Wqb = (__bf16*)carve((size_t)QDIM * HIDDEN * 2);   //  8 MiB
    __bf16* Wkb = (__bf16*)carve((size_t)KVDIM * HIDDEN * 2);  //  4 MiB
    __bf16* Wvb = (__bf16*)carve((size_t)KVDIM * HIDDEN * 2);  //  4 MiB
    __bf16* Wob = (__bf16*)carve((size_t)HIDDEN * QDIM * 2);   //  8 MiB
    float*  Qf  = (float*) carve((size_t)NTOK * QDIM * 4);     // 32 MiB
    float*  Kf  = (float*) carve((size_t)NTOK * KVDIM * 4);    // 16 MiB
    float*  Vf  = (float*) carve((size_t)NTOK * KVDIM * 4);    // 16 MiB
    __bf16* Qb  = (__bf16*)carve((size_t)NTOK * QDIM * 2);     // 16 MiB
    __bf16* Kb  = (__bf16*)carve((size_t)NTOK * KVDIM * 2);    //  8 MiB
    __bf16* Vb  = (__bf16*)carve((size_t)NTOK * KVDIM * 2);    //  8 MiB
    __bf16* ctx = (__bf16*)Qf;  // alias: Qf is dead after norm_rope_kernel

    auto cvt = [&](const float* s, __bf16* d, long n) {
        cvt_f32_bf16_kernel<<<(int)((n + 255) / 256), 256, 0, stream>>>(s, d, n);
    };
    cvt(hidden, Xb,  (long)NTOK * HIDDEN);
    cvt(Wq,     Wqb, (long)QDIM * HIDDEN);
    cvt(Wk,     Wkb, (long)KVDIM * HIDDEN);
    cvt(Wv,     Wvb, (long)KVDIM * HIDDEN);
    cvt(Wo,     Wob, (long)HIDDEN * QDIM);

    // Q/K/V projections: C = X * W^T
    {
        dim3 blk(256);
        gemm_bf16_nt_kernel<<<dim3(QDIM / 128,  NTOK / 64), blk, 0, stream>>>(
            Xb, Wqb, Qf, NTOK, QDIM, HIDDEN);
        gemm_bf16_nt_kernel<<<dim3(KVDIM / 128, NTOK / 64), blk, 0, stream>>>(
            Xb, Wkb, Kf, NTOK, KVDIM, HIDDEN);
        gemm_bf16_nt_kernel<<<dim3(KVDIM / 128, NTOK / 64), blk, 0, stream>>>(
            Xb, Wvb, Vf, NTOK, KVDIM, HIDDEN);
    }

    // RMSNorm + RoPE (+ bf16 repack of V)
    norm_rope_kernel<<<NTOK * 32, 128, 0, stream>>>(
        Qf, Kf, Vf, cosb, sinb, qw, kw, Qb, Kb, Vb);

    // Fused causal attention: probs -> attn_out, context -> ctx (bf16)
    attn_kernel<<<dim3(SEQ / 16, BATCH * NH), 256, 0, stream>>>(
        Qb, Kb, Vb, attn_out, ctx);

    // Output projection: out = ctx * Wo^T
    gemm_bf16_nt_kernel<<<dim3(HIDDEN / 128, NTOK / 64), 256, 0, stream>>>(
        ctx, Wob, out_proj, NTOK, HIDDEN, QDIM);
}
